// MultivariateTSModel_6124623364316
// MI455X (gfx1250) — compile-verified, matrix operand order checked by reading, not run on hardware
//
#include <hip/hip_runtime.h>
#include <hip/hip_bf16.h>

typedef __attribute__((ext_vector_type(16))) _Float16 v16h;
typedef __attribute__((ext_vector_type(8)))  float    v8f;

#define B_   64
#define T_   200
#define C_   53
#define EMB_ 64
#define HID_ 128
#define C2_  2809
#define C2P  2816
#define D1_  140
#define D1P  160
#define D2_  1404
#define D2P  1408
#define H2N  702
#define H2P  704

__device__ __forceinline__ float sigm(float x) { return 1.f / (1.f + __expf(-x)); }

union FragU { uint4 u[2]; v16h h; };

// Load a 16x32 f16 WMMA fragment from a row-major [rows][ld] array.
// A-fragment: rowbase = tile row base (M).  B-fragment: array stored [N][K], rowbase = tile col base (N).
// Lane layout per CDNA5 ISA: lane m=lane&15, half=lane>>4; K runs [half*8,+8) and [16+half*8,+8).
__device__ __forceinline__ v16h ldfrag(const _Float16* base, int ld, int rowbase, int k0) {
  int lane = threadIdx.x & 31;
  int hl = lane >> 4;
  const _Float16* p = base + (size_t)(rowbase + (lane & 15)) * ld + k0 + hl * 8;
  FragU f;
  f.u[0] = *(const uint4*)(p);
  f.u[1] = *(const uint4*)(p + 16);
  return f.h;
}

__device__ __forceinline__ v8f wmma32(v16h a, v16h b, v8f c) {
  return __builtin_amdgcn_wmma_f32_16x16x32_f16(false, a, false, b, (short)0, c, false, false);
}

// Build an A-fragment of the per-scalar embedding e[c][kk] = x_c*emb_w[kk]+emb_b[kk]
// in registers (no LDS staging).  bl = bias_lds (emb_w at [0,64), emb_b at [64,128)).
__device__ __forceinline__ v16h efrag_make(float xc, const float* bl, int ks, int hl) {
  v16h a;
  #pragma unroll
  for (int hh = 0; hh < 2; ++hh) {
    int kb = ks + hh * 16 + hl * 8;
    #pragma unroll
    for (int e = 0; e < 8; ++e) {
      int kk = kb + e;
      a[hh * 8 + e] = (_Float16)(xc * bl[kk] + bl[64 + kk]);
    }
  }
  return a;
}

// ---------------------------------------------------------------------------
// Weight convert + transpose:  src f32 [K][N]  ->  dst f16 [Npad][Kpad], zero padded
// ---------------------------------------------------------------------------
__global__ void convT_kernel(const float* __restrict__ src, _Float16* __restrict__ dst,
                             int K, int N, int Kpad, int Npad) {
  int idx = blockIdx.x * 256 + threadIdx.x;
  int tot = Npad * Kpad;
  if (idx >= tot) return;
  int n = idx / Kpad, k = idx - n * Kpad;
  float v = (n < N && k < K) ? src[(size_t)k * N + n] : 0.f;
  dst[idx] = (_Float16)v;
}

// ---------------------------------------------------------------------------
// Recurrent kernel: one block per batch element, loops over all T steps.
// blockDim = 256 (8 waves).  W_hh + Q/K/V weights LDS-resident (loaded once);
// h double-buffered in LDS; gates/qkv split into 2 passes to cap VGPR pressure.
// Total static LDS ~302KB (<= 320KB/WG per CDNA5 ISA) -> 1 workgroup per WGP.
// ---------------------------------------------------------------------------
__global__ __launch_bounds__(256) void rnn_kernel(
    const float* __restrict__ x, const float* __restrict__ emb_w, const float* __restrict__ emb_b,
    const float* __restrict__ b_ih, const float* __restrict__ b_hh,
    const float* __restrict__ q_b, const float* __restrict__ k_b, const float* __restrict__ v_b,
    const float* __restrict__ gate_b,
    const _Float16* __restrict__ WIH, const _Float16* __restrict__ WHH,
    const _Float16* __restrict__ QW, const _Float16* __restrict__ KW, const _Float16* __restrict__ VW,
    float* __restrict__ align_out)
{
  __shared__ alignas(16) _Float16 WHH_l[384][128];  // 96KB  [N][K]
  __shared__ alignas(16) _Float16 QW_l[128][128];   // 32KB
  __shared__ alignas(16) _Float16 KW_l[128][128];   // 32KB
  __shared__ alignas(16) _Float16 VW_l[128][128];   // 32KB
  __shared__ alignas(16) _Float16 HA[64][128];      // h state
  __shared__ alignas(16) _Float16 HB[64][128];      // GRU output (attn input)
  __shared__ alignas(16) _Float16 q_lds[64][128];
  __shared__ alignas(16) _Float16 k_lds[64][128];   // row-major k == B operand of q@k^T
  __shared__ alignas(16) _Float16 vT_lds[128][64];  // v transposed == B operand of attn@v
  __shared__ alignas(16) float    sc_lds[64][64];
  __shared__ alignas(16) _Float16 at_lds[64][64];
  __shared__ float bias_lds[1280]; // emb_w,emb_b,b_ih,b_hh,q_b,k_b,v_b
  __shared__ float pv[64], pt[64], red[2];

  const int b    = blockIdx.x;
  const int tid  = threadIdx.x;
  const int wv   = tid >> 5;
  const int lane = tid & 31;
  const int ml   = lane & 15;
  const int hl   = lane >> 4;

  // one-time LDS preload of weights (wide copies; f16 sources already transposed)
  {
    const uint4* s = (const uint4*)WHH; uint4* d = (uint4*)&WHH_l[0][0];
    for (int i = tid; i < 384 * 128 / 8; i += 256) d[i] = s[i];
    const uint4* sq = (const uint4*)QW; uint4* dq = (uint4*)&QW_l[0][0];
    const uint4* sk = (const uint4*)KW; uint4* dk = (uint4*)&KW_l[0][0];
    const uint4* sv = (const uint4*)VW; uint4* dv = (uint4*)&VW_l[0][0];
    for (int i = tid; i < 128 * 128 / 8; i += 256) { dq[i] = sq[i]; dk[i] = sk[i]; dv[i] = sv[i]; }
  }
  for (int i = tid; i < 64; i += 256)  { bias_lds[i] = emb_w[i]; bias_lds[64 + i] = emb_b[i]; }
  for (int i = tid; i < 384; i += 256) { bias_lds[128 + i] = b_ih[i]; bias_lds[512 + i] = b_hh[i]; }
  for (int i = tid; i < 128; i += 256) { bias_lds[896 + i] = q_b[i]; bias_lds[1024 + i] = k_b[i]; bias_lds[1152 + i] = v_b[i]; }
  for (int i = tid; i < 64 * 128; i += 256) ((_Float16*)HA)[i] = (_Float16)0.f;
  __syncthreads();

  const int m  = wv >> 1;        // row tile 0..3 (16 rows each, rows 53..63 padded)
  const int cb = (wv & 1) * 4;   // col tile base out of 8 (HID=128)
  const int crow = m * 16 + ml;  // component row this lane's A-fragments cover
  const v8f zf = {};

  for (int t = 0; t < T_; ++t) {
    float xc = (crow < C_) ? x[((size_t)b * T_ + t) * C_ + crow] : 0.f;

    // ---- GRU gates: gi = e@Wih (global B), gh = h@Whh (LDS B); write h_gru -> HB ----
    #pragma unroll
    for (int p = 0; p < 2; ++p) {
      v8f aR[2], aZ[2], aNi[2], aNh[2];
      aR[0] = zf; aR[1] = zf; aZ[0] = zf; aZ[1] = zf;
      aNi[0] = zf; aNi[1] = zf; aNh[0] = zf; aNh[1] = zf;
      #pragma unroll
      for (int ks = 0; ks < 64; ks += 32) {
        v16h a = efrag_make(xc, bias_lds, ks, hl);
        #pragma unroll
        for (int j = 0; j < 2; ++j) {
          int ct = cb + p * 2 + j;
          aR[j]  = wmma32(a, ldfrag(WIH, 64, ct * 16,       ks), aR[j]);
          aZ[j]  = wmma32(a, ldfrag(WIH, 64, 128 + ct * 16, ks), aZ[j]);
          aNi[j] = wmma32(a, ldfrag(WIH, 64, 256 + ct * 16, ks), aNi[j]);
        }
      }
      #pragma unroll
      for (int ks = 0; ks < 128; ks += 32) {
        v16h a = ldfrag(&HA[0][0], 128, m * 16, ks);
        #pragma unroll
        for (int j = 0; j < 2; ++j) {
          int ct = cb + p * 2 + j;
          aR[j]  = wmma32(a, ldfrag(&WHH_l[0][0], 128, ct * 16,       ks), aR[j]);
          aZ[j]  = wmma32(a, ldfrag(&WHH_l[0][0], 128, 128 + ct * 16, ks), aZ[j]);
          aNh[j] = wmma32(a, ldfrag(&WHH_l[0][0], 128, 256 + ct * 16, ks), aNh[j]);
        }
      }
      #pragma unroll
      for (int j = 0; j < 2; ++j) {
        int cidx = (cb + p * 2 + j) * 16 + ml;
        float br  = bias_lds[128 + cidx] + bias_lds[512 + cidx];
        float bz  = bias_lds[256 + cidx] + bias_lds[640 + cidx];
        float bin = bias_lds[384 + cidx];
        float bhn = bias_lds[768 + cidx];
        #pragma unroll
        for (int r = 0; r < 8; ++r) {
          int i = m * 16 + r + hl * 8;
          float rr = sigm(aR[j][r] + br);
          float zz = sigm(aZ[j][r] + bz);
          float nn = tanhf(aNi[j][r] + bin + rr * (aNh[j][r] + bhn));
          float ho = (float)HA[i][cidx];
          HB[i][cidx] = (_Float16)((1.f - zz) * nn + zz * ho);
        }
      }
    }
    __syncthreads();

    // ---- q, k, v projections from HB (all B operands LDS-resident) ----
    #pragma unroll
    for (int p = 0; p < 2; ++p) {
      v8f qa[2], ka[2], va[2];
      qa[0] = zf; qa[1] = zf; ka[0] = zf; ka[1] = zf; va[0] = zf; va[1] = zf;
      #pragma unroll
      for (int ks = 0; ks < 128; ks += 32) {
        v16h a = ldfrag(&HB[0][0], 128, m * 16, ks);
        #pragma unroll
        for (int j = 0; j < 2; ++j) {
          int ct = cb + p * 2 + j;
          qa[j] = wmma32(a, ldfrag(&QW_l[0][0], 128, ct * 16, ks), qa[j]);
          ka[j] = wmma32(a, ldfrag(&KW_l[0][0], 128, ct * 16, ks), ka[j]);
          va[j] = wmma32(a, ldfrag(&VW_l[0][0], 128, ct * 16, ks), va[j]);
        }
      }
      #pragma unroll
      for (int j = 0; j < 2; ++j) {
        int n = (cb + p * 2 + j) * 16 + ml;
        #pragma unroll
        for (int r = 0; r < 8; ++r) {
          int i = m * 16 + r + hl * 8;
          q_lds[i][n]  = (_Float16)(qa[j][r] + bias_lds[896 + n]);
          k_lds[i][n]  = (_Float16)(ka[j][r] + bias_lds[1024 + n]);
          vT_lds[n][i] = (_Float16)(va[j][r] + bias_lds[1152 + n]);
        }
      }
    }
    __syncthreads();

    // ---- scores = tanh(q @ k^T), zero padded outside 53x53 ----
    {
      v8f sa[2]; sa[0] = zf; sa[1] = zf;
      int c2b = (wv & 1) * 2;
      #pragma unroll
      for (int ks = 0; ks < 128; ks += 32) {
        v16h a = ldfrag(&q_lds[0][0], 128, m * 16, ks);
        #pragma unroll
        for (int j = 0; j < 2; ++j)
          sa[j] = wmma32(a, ldfrag(&k_lds[0][0], 128, (c2b + j) * 16, ks), sa[j]);
      }
      #pragma unroll
      for (int j = 0; j < 2; ++j) {
        int jc = (c2b + j) * 16 + ml;
        #pragma unroll
        for (int r = 0; r < 8; ++r) {
          int i = m * 16 + r + hl * 8;
          sc_lds[i][jc] = (i < C_ && jc < C_) ? tanhf(sa[j][r]) : 0.f;
        }
      }
    }
    __syncthreads();

    // ---- spectral norm via power iteration on S^T S ----
    if (tid < 64) pv[tid] = (tid < C_) ? 1.f : 0.f;
    __syncthreads();
    for (int it = 0; it < 12; ++it) {
      if (tid < C_) { float s = 0.f; for (int j2 = 0; j2 < C_; ++j2) s += sc_lds[tid][j2] * pv[j2]; pt[tid] = s; }
      __syncthreads();
      if (tid < C_) { float s = 0.f; for (int i2 = 0; i2 < C_; ++i2) s += sc_lds[i2][tid] * pt[i2]; pv[tid] = s; }
      __syncthreads();
      if (tid == 0) { float s = 0.f; for (int j2 = 0; j2 < C_; ++j2) s += pv[j2] * pv[j2]; red[0] = rsqrtf(fmaxf(s, 1e-30f)); }
      __syncthreads();
      if (tid < C_) pv[tid] *= red[0];
      __syncthreads();
    }
    if (tid < C_) { float s = 0.f; for (int j2 = 0; j2 < C_; ++j2) s += sc_lds[tid][j2] * pv[j2]; pt[tid] = s; }
    __syncthreads();
    if (tid == 0) { float s = 0.f; for (int i2 = 0; i2 < C_; ++i2) s += pt[i2] * pt[i2]; red[1] = 1.f / fmaxf(sqrtf(s), 1e-12f); }
    __syncthreads();
    float invs = red[1];

    // ---- attn = s * sigmoid(|s| + gate_b); write align; stage f16 (zero padded) ----
    for (int idx = tid; idx < 64 * 64; idx += 256) {
      int i = idx >> 6, jc = idx & 63;
      float a = 0.f;
      if (i < C_ && jc < C_) {
        float s = sc_lds[i][jc] * invs;
        a = s * sigm(fabsf(s) + gate_b[i * C_ + jc]);
        align_out[(((size_t)b * T_ + t) * C_ + i) * C_ + jc] = a;
      }
      at_lds[i][jc] = (_Float16)a;
    }
    __syncthreads();

    // ---- h_state = attn @ v  -> HA ----
    {
      v8f ha[4];
      #pragma unroll
      for (int j = 0; j < 4; ++j) ha[j] = zf;
      #pragma unroll
      for (int ks = 0; ks < 64; ks += 32) {
        v16h a = ldfrag(&at_lds[0][0], 64, m * 16, ks);
        #pragma unroll
        for (int j = 0; j < 4; ++j)
          ha[j] = wmma32(a, ldfrag(&vT_lds[0][0], 64, (cb + j) * 16, ks), ha[j]);
      }
      #pragma unroll
      for (int j = 0; j < 4; ++j) {
        int n = (cb + j) * 16 + ml;
        #pragma unroll
        for (int r = 0; r < 8; ++r) HA[m * 16 + r + hl * 8][n] = (_Float16)ha[j][r];
      }
    }
    __syncthreads();
  }
}

// ---------------------------------------------------------------------------
// am[b][k] = mean_t align[b][t][k]   (f32, zero padded to C2P)
// ---------------------------------------------------------------------------
__global__ void am_kernel(const float* __restrict__ align, float* __restrict__ am) {
  int c2 = blockIdx.x * 256 + threadIdx.x;
  int b = blockIdx.y;
  if (c2 >= C2P) return;
  float s = 0.f;
  if (c2 < C2_) {
    for (int t = 0; t < T_; ++t) s += align[((size_t)b * T_ + t) * C2_ + c2];
    s *= (1.f / T_);
  }
  am[(size_t)b * C2P + c2] = s;
}

// ---------------------------------------------------------------------------
// xe = (ai * am) @ ge_w + ge_b : M=12800, K=2809(pad 2816), N=140(pad 144)
// ---------------------------------------------------------------------------
__global__ __launch_bounds__(128) void gemm_xe(const float* __restrict__ align,
    const float* __restrict__ am, const _Float16* __restrict__ GEW,
    const float* __restrict__ ge_b, float* __restrict__ xe) {
  int wv = threadIdx.x >> 5, lane = threadIdx.x & 31, ml = lane & 15, hl = lane >> 4;
  int rowb = blockIdx.x * 64 + wv * 16;
  int row = rowb + ml;
  const float* ar  = align + (size_t)row * C2_;
  const float* amr = am + (size_t)(row / T_) * C2P;
  v8f acc[9]; const v8f zf = {};
  #pragma unroll
  for (int j = 0; j < 9; ++j) acc[j] = zf;
  for (int ks = 0; ks < C2P; ks += 32) {
    if (ks + 64 < C2_) __builtin_prefetch(ar + ks + 64, 0, 0);
    v16h a;
    #pragma unroll
    for (int hh = 0; hh < 2; ++hh) {
      int kb = ks + hh * 16 + hl * 8;
      #pragma unroll
      for (int e = 0; e < 8; ++e) {
        int k = kb + e;
        float v = (k < C2_) ? ar[k] * amr[k] : 0.f;
        a[hh * 8 + e] = (_Float16)v;
      }
    }
    #pragma unroll
    for (int j = 0; j < 9; ++j) acc[j] = wmma32(a, ldfrag(GEW, C2P, j * 16, ks), acc[j]);
  }
  #pragma unroll
  for (int j = 0; j < 9; ++j) {
    int n = j * 16 + ml;
    float bb = (n < D1_) ? ge_b[n] : 0.f;
    #pragma unroll
    for (int r = 0; r < 8; ++r)
      xe[(size_t)(rowb + r + hl * 8) * 144 + n] = acc[j][r] + bb;
  }
}

// BN stats: per-column scale/shift (training-mode biased var)
__global__ __launch_bounds__(256) void bn_stats(const float* __restrict__ xe,
    const float* __restrict__ g, const float* __restrict__ bb, float* __restrict__ st) {
  __shared__ float s1[256], s2[256];
  int c = blockIdx.x;
  float a = 0.f, q = 0.f;
  for (int r = threadIdx.x; r < B_ * T_; r += 256) { float v = xe[(size_t)r * 144 + c]; a += v; q += v * v; }
  s1[threadIdx.x] = a; s2[threadIdx.x] = q;
  __syncthreads();
  for (int off = 128; off > 0; off >>= 1) {
    if (threadIdx.x < off) { s1[threadIdx.x] += s1[threadIdx.x + off]; s2[threadIdx.x] += s2[threadIdx.x + off]; }
    __syncthreads();
  }
  if (threadIdx.x == 0) {
    float mean = s1[0] / (B_ * T_);
    float var  = s2[0] / (B_ * T_) - mean * mean;
    float sc = rsqrtf(var + 1e-5f) * g[c];
    st[c * 2] = sc;
    st[c * 2 + 1] = bb[c] - mean * sc;
  }
}

__global__ void xn_kernel(const float* __restrict__ xe, const float* __restrict__ st,
                          _Float16* __restrict__ xn) {
  int idx = blockIdx.x * 256 + threadIdx.x;
  int row = idx / D1P, c = idx - row * D1P;
  float v = 0.f;
  if (c < D1_) v = fmaxf(xe[(size_t)row * 144 + c] * st[c * 2] + st[c * 2 + 1], 0.f);
  xn[idx] = (_Float16)v;
}

// ---------------------------------------------------------------------------
// Fused: a = relu(xn @ ga_w1 + b1) @ ga_w2 + b2    (avoids 72MB intermediate)
// ---------------------------------------------------------------------------
__global__ __launch_bounds__(128) void gemm_att(const _Float16* __restrict__ xn,
    const _Float16* __restrict__ GA1, const float* __restrict__ gb1,
    const float* __restrict__ gw2, const float* __restrict__ gb2,
    float* __restrict__ avec) {
  int wv = threadIdx.x >> 5, lane = threadIdx.x & 31, ml = lane & 15, hl = lane >> 4;
  int rowb = blockIdx.x * 64 + wv * 16;
  v16h a5[5];
  #pragma unroll
  for (int ks = 0; ks < 5; ++ks) a5[ks] = ldfrag(xn, D1P, rowb, ks * 32);
  float racc[8];
  #pragma unroll
  for (int r = 0; r < 8; ++r) racc[r] = 0.f;
  const v8f zf = {};
  for (int ct = 0; ct < 88; ++ct) {
    v8f acc = zf;
    #pragma unroll
    for (int ks = 0; ks < 5; ++ks) acc = wmma32(a5[ks], ldfrag(GA1, D1P, ct * 16, ks * 32), acc);
    int n = ct * 16 + ml;
    float b1 = (n < D2_) ? gb1[n] : 0.f;
    float g2 = (n < D2_) ? gw2[n] : 0.f;
    #pragma unroll
    for (int r = 0; r < 8; ++r) racc[r] += fmaxf(acc[r] + b1, 0.f) * g2;
  }
  #pragma unroll
  for (int r = 0; r < 8; ++r) {
    float v = racc[r];
    #pragma unroll
    for (int off = 1; off < 16; off <<= 1) v += __shfl_xor(v, off, 16);
    if (ml == 0) avec[rowb + r + hl * 8] = v + gb2[0];
  }
}

__global__ __launch_bounds__(256) void softmax_kernel(const float* __restrict__ avec, float* __restrict__ w) {
  __shared__ float sm[256];
  int b = blockIdx.x;
  float v = (threadIdx.x < T_) ? avec[b * T_ + threadIdx.x] : -1e30f;
  sm[threadIdx.x] = v;
  __syncthreads();
  for (int off = 128; off > 0; off >>= 1) { if (threadIdx.x < off) sm[threadIdx.x] = fmaxf(sm[threadIdx.x], sm[threadIdx.x + off]); __syncthreads(); }
  float mx = sm[0];
  __syncthreads();
  float e = (threadIdx.x < T_) ? __expf(v - mx) : 0.f;
  sm[threadIdx.x] = e;
  __syncthreads();
  for (int off = 128; off > 0; off >>= 1) { if (threadIdx.x < off) sm[threadIdx.x] += sm[threadIdx.x + off]; __syncthreads(); }
  if (threadIdx.x < T_) w[b * T_ + threadIdx.x] = e / sm[0];
}

__global__ void dnc_kernel(const float* __restrict__ align, const float* __restrict__ w,
                           float* __restrict__ dnc, _Float16* __restrict__ dncf) {
  int c2 = blockIdx.x * 256 + threadIdx.x;
  int b = blockIdx.y;
  if (c2 >= C2P) return;
  float s = 0.f;
  if (c2 < C2_) {
    for (int t = 0; t < T_; ++t) s += align[((size_t)b * T_ + t) * C2_ + c2] * w[b * T_ + t];
    dnc[(size_t)b * C2_ + c2] = s;
  }
  dncf[(size_t)b * C2P + c2] = (_Float16)s;
}

// classifier layer 1: h1 = relu(dnc @ c_w1 + b1), M=64 K=2816 N=1408
__global__ __launch_bounds__(128) void gemm_h1(const _Float16* __restrict__ dncf,
    const _Float16* __restrict__ CW1, const float* __restrict__ cb1, _Float16* __restrict__ h1) {
  int wv = threadIdx.x >> 5, lane = threadIdx.x & 31, ml = lane & 15, hl = lane >> 4;
  v8f acc[4]; const v8f zf = {};
  #pragma unroll
  for (int j = 0; j < 4; ++j) acc[j] = zf;
  for (int ks = 0; ks < C2P; ks += 32) {
    if (ks + 64 < C2P) __builtin_prefetch(CW1 + (size_t)(blockIdx.x * 4) * 16 * C2P + ks + 64, 0, 0);
    v16h a = ldfrag(dncf, C2P, wv * 16, ks);
    #pragma unroll
    for (int j = 0; j < 4; ++j)
      acc[j] = wmma32(a, ldfrag(CW1, C2P, (blockIdx.x * 4 + j) * 16, ks), acc[j]);
  }
  #pragma unroll
  for (int j = 0; j < 4; ++j) {
    int n = (blockIdx.x * 4 + j) * 16 + ml;
    float bb = (n < D2_) ? cb1[n] : 0.f;
    #pragma unroll
    for (int r = 0; r < 8; ++r) {
      float v = (n < D2_) ? fmaxf(acc[j][r] + bb, 0.f) : 0.f;
      h1[(size_t)(wv * 16 + r + hl * 8) * D2P + n] = (_Float16)v;
    }
  }
}

// classifier layer 2: h2 = relu(h1 @ c_w2 + b2), M=64 K=1408 N=704
__global__ __launch_bounds__(128) void gemm_h2(const _Float16* __restrict__ h1,
    const _Float16* __restrict__ CW2, const float* __restrict__ cb2, _Float16* __restrict__ h2) {
  int wv = threadIdx.x >> 5, lane = threadIdx.x & 31, ml = lane & 15, hl = lane >> 4;
  v8f acc[4]; const v8f zf = {};
  #pragma unroll
  for (int j = 0; j < 4; ++j) acc[j] = zf;
  for (int ks = 0; ks < D2P; ks += 32) {
    v16h a = ldfrag(h1, D2P, wv * 16, ks);
    #pragma unroll
    for (int j = 0; j < 4; ++j)
      acc[j] = wmma32(a, ldfrag(CW2, D2P, (blockIdx.x * 4 + j) * 16, ks), acc[j]);
  }
  #pragma unroll
  for (int j = 0; j < 4; ++j) {
    int n = (blockIdx.x * 4 + j) * 16 + ml;
    float bb = (n < H2N) ? cb2[n] : 0.f;
    #pragma unroll
    for (int r = 0; r < 8; ++r) {
      float v = (n < H2N) ? fmaxf(acc[j][r] + bb, 0.f) : 0.f;
      h2[(size_t)(wv * 16 + r + hl * 8) * H2P + n] = (_Float16)v;
    }
  }
}

__global__ __launch_bounds__(128) void logits_kernel(const _Float16* __restrict__ h2,
    const float* __restrict__ w3, const float* __restrict__ b3, float* __restrict__ out) {
  int row = threadIdx.x >> 1, o = threadIdx.x & 1;
  float s = b3[o];
  for (int k = 0; k < H2N; ++k) s += (float)h2[(size_t)row * H2P + k] * w3[k * 2 + o];
  out[row * 2 + o] = s;
}

// ---------------------------------------------------------------------------
// Workspace layout (bytes, 256-aligned)
// ---------------------------------------------------------------------------
static const size_t OFF_WIH = 0;         // 384x64  f16
static const size_t OFF_WHH = 49152;     // 384x128 f16
static const size_t OFF_QW  = 147456;    // 128x128 f16
static const size_t OFF_KW  = 180224;
static const size_t OFF_VW  = 212992;
static const size_t OFF_GEW = 245760;    // 144x2816 f16
static const size_t OFF_GA1 = 1056768;   // 1408x160 f16
static const size_t OFF_CW1 = 1507328;   // 1408x2816 f16
static const size_t OFF_CW2 = 9437184;   // 704x1408 f16
static const size_t OFF_AM  = 11419648;  // 64x2816 f32
static const size_t OFF_XE  = 12140544;  // 12800x144 f32
static const size_t OFF_XN  = 19513344;  // 12800x160 f16
static const size_t OFF_BN  = 23609344;  // 140x2 f32
static const size_t OFF_AV  = 23610624;  // 12800 f32
static const size_t OFF_SM  = 23661824;  // 64x200 f32
static const size_t OFF_DNC = 23713024;  // 64x2816 f16
static const size_t OFF_H1  = 24073472;  // 64x1408 f16
static const size_t OFF_H2  = 24253696;  // 64x704 f16

extern "C" void kernel_launch(void* const* d_in, const int* in_sizes, int n_in,
                              void* d_out, int out_size, void* d_ws, size_t ws_size,
                              hipStream_t stream) {
  const float* x     = (const float*)d_in[0];
  const float* emb_w = (const float*)d_in[1];
  const float* emb_b = (const float*)d_in[2];
  const float* w_ih  = (const float*)d_in[3];
  const float* w_hh  = (const float*)d_in[4];
  const float* b_ih  = (const float*)d_in[5];
  const float* b_hh  = (const float*)d_in[6];
  const float* q_w   = (const float*)d_in[7];
  const float* q_b   = (const float*)d_in[8];
  const float* k_w   = (const float*)d_in[9];
  const float* k_b   = (const float*)d_in[10];
  const float* v_w   = (const float*)d_in[11];
  const float* v_b   = (const float*)d_in[12];
  const float* gate_b= (const float*)d_in[13];
  const float* ge_w  = (const float*)d_in[14];
  const float* ge_b  = (const float*)d_in[15];
  const float* bn_g  = (const float*)d_in[16];
  const float* bn_b  = (const float*)d_in[17];
  const float* ga_w1 = (const float*)d_in[18];
  const float* ga_b1 = (const float*)d_in[19];
  const float* ga_w2 = (const float*)d_in[20];
  const float* ga_b2 = (const float*)d_in[21];
  const float* c_w1  = (const float*)d_in[22];
  const float* c_b1  = (const float*)d_in[23];
  const float* c_w2  = (const float*)d_in[24];
  const float* c_b2  = (const float*)d_in[25];
  const float* c_w3  = (const float*)d_in[26];
  const float* c_b3  = (const float*)d_in[27];

  float* out    = (float*)d_out;
  float* logits = out;
  float* dnc    = out + 128;
  float* align  = out + 128 + (size_t)B_ * C2_;

  char* ws = (char*)d_ws;
  _Float16* WIH  = (_Float16*)(ws + OFF_WIH);
  _Float16* WHH  = (_Float16*)(ws + OFF_WHH);
  _Float16* QW   = (_Float16*)(ws + OFF_QW);
  _Float16* KW   = (_Float16*)(ws + OFF_KW);
  _Float16* VW   = (_Float16*)(ws + OFF_VW);
  _Float16* GEW  = (_Float16*)(ws + OFF_GEW);
  _Float16* GA1  = (_Float16*)(ws + OFF_GA1);
  _Float16* CW1  = (_Float16*)(ws + OFF_CW1);
  _Float16* CW2  = (_Float16*)(ws + OFF_CW2);
  float*    AM   = (float*)(ws + OFF_AM);
  float*    XE   = (float*)(ws + OFF_XE);
  _Float16* XN   = (_Float16*)(ws + OFF_XN);
  float*    BN   = (float*)(ws + OFF_BN);
  float*    AV   = (float*)(ws + OFF_AV);
  float*    SMW  = (float*)(ws + OFF_SM);
  _Float16* DNCF = (_Float16*)(ws + OFF_DNC);
  _Float16* H1F  = (_Float16*)(ws + OFF_H1);
  _Float16* H2F  = (_Float16*)(ws + OFF_H2);

  // f16 transposed ([N][Kpad]) weight copies
  auto cvt = [&](const float* src, _Float16* dst, int K, int N, int Kpad, int Npad) {
    int tot = Npad * Kpad;
    convT_kernel<<<(tot + 255) / 256, 256, 0, stream>>>(src, dst, K, N, Kpad, Npad);
  };
  cvt(w_ih, WIH, 64, 384, 64, 384);
  cvt(w_hh, WHH, 128, 384, 128, 384);
  cvt(q_w, QW, 128, 128, 128, 128);
  cvt(k_w, KW, 128, 128, 128, 128);
  cvt(v_w, VW, 128, 128, 128, 128);
  cvt(ge_w, GEW, C2_, D1_, C2P, 144);
  cvt(ga_w1, GA1, D1_, D2_, D1P, D2P);
  cvt(c_w1, CW1, C2_, D2_, C2P, D2P);
  cvt(c_w2, CW2, D2_, H2N, D2P, H2P);

  // GRU + attention recurrence: one block per batch element, loops over T
  rnn_kernel<<<B_, 256, 0, stream>>>(x, emb_w, emb_b, b_ih, b_hh, q_b, k_b, v_b,
                                     gate_b, WIH, WHH, QW, KW, VW, align);

  // gta attention over time
  am_kernel<<<dim3(11, 64), 256, 0, stream>>>(align, AM);
  gemm_xe<<<200, 128, 0, stream>>>(align, AM, GEW, ge_b, XE);
  bn_stats<<<140, 256, 0, stream>>>(XE, bn_g, bn_b, BN);
  xn_kernel<<<(12800 * D1P) / 256, 256, 0, stream>>>(XE, BN, XN);
  gemm_att<<<200, 128, 0, stream>>>(XN, GA1, ga_b1, ga_w2, ga_b2, AV);
  softmax_kernel<<<64, 256, 0, stream>>>(AV, SMW);
  dnc_kernel<<<dim3(11, 64), 256, 0, stream>>>(align, SMW, dnc, DNCF);

  // classifier
  gemm_h1<<<22, 128, 0, stream>>>(DNCF, CW1, c_b1, H1F);
  gemm_h2<<<11, 128, 0, stream>>>(H1F, CW2, c_b2, H2F);
  logits_kernel<<<1, 128, 0, stream>>>(H2F, c_w3, c_b3, logits);
}
